// RNNODEEncoder_17317308138021
// MI455X (gfx1250) — matrix operand order repaired and checked by reading, not compiled, not run on hardware
//
#include <hip/hip_runtime.h>
#include <hip/hip_bf16.h>
#include <math.h>

typedef __attribute__((ext_vector_type(16))) _Float16 v16h;
typedef __attribute__((ext_vector_type(8)))  _Float16 v8h;
typedef __attribute__((ext_vector_type(8)))  float    v8f;

#define B_SZ   2048
#define T_SZ   512
#define D_IN   64
#define H_SZ   128
#define D_OUT  64
#define YS     136   // padded row stride (halfs) for activation staging
#define WS     136   // padded row stride (halfs) for weights

// ---- fast hardware transcendentals (latency chain is the bottleneck) ----
__device__ __forceinline__ float fast_rcp(float a) {
#if __has_builtin(__builtin_amdgcn_rcpf)
    return __builtin_amdgcn_rcpf(a);          // v_rcp_f32
#else
    return 1.0f / a;
#endif
}

__device__ __forceinline__ float fast_tanh(float a) {
#if __has_builtin(__builtin_amdgcn_tanhf)
    return __builtin_amdgcn_tanhf(a);         // v_tanh_f32 (gfx1250 trans op)
#elif __has_builtin(__builtin_amdgcn_tanh_f32)
    return __builtin_amdgcn_tanh_f32(a);
#else
    // tanh(x) = 1 - 2/(exp(2x)+1), via v_exp_f32 + v_rcp_f32
    return 1.0f - 2.0f * fast_rcp(__expf(2.0f * a) + 1.0f);
#endif
}

__device__ __forceinline__ float fast_silu(float s) {
    // s * sigmoid(s) via v_exp_f32 + v_rcp_f32 (avoid IEEE div sequence)
    return s * fast_rcp(1.0f + __expf(-s));
}

struct alignas(16) Smem {
    _Float16 whh[H_SZ * WS];   // W_hh  (f16, row-major [n][k], padded)
    _Float16 w1 [H_SZ * WS];   // W1
    _Float16 w2 [H_SZ * WS];   // W2
    _Float16 ybuf[16 * YS];    // activation staging tile [16 rows][128 cols]
    float    hbuf[16 * H_SZ];  // final h (f32) for output epilogue
    float    we[H_SZ];         // W_ih @ W_emb
    float    cc[H_SZ];         // W_ih @ b_emb + b_ih + b_hh
    float    xbuf[16];         // x values for current step
};

// GEMM tile: c[2] (+)= y[16,128] @ W^T for this wave's two 16-col blocks.
// ybuf row-major f16 [16][YS]; wl row-major f16 [128][WS] where wl[n][k] = W[n,k].
__device__ __forceinline__ void gemm8(const _Float16* __restrict__ yb,
                                      const _Float16* __restrict__ wl,
                                      int lane, int w, v8f c[2]) {
    const int nl = lane & 15;
    const int hi = lane >> 4;
    v16h A[4];
#pragma unroll
    for (int kk = 0; kk < 4; ++kk) {
        union { v16h v; v8h h[2]; } u;
        const _Float16* p = yb + nl * YS + kk * 32 + hi * 8;
        u.h[0] = *(const v8h*)(p);        // K = kk*32 + hi*8  .. +7
        u.h[1] = *(const v8h*)(p + 16);   // K = kk*32 + 16 + hi*8 .. +7
        A[kk] = u.v;
    }
#pragma unroll
    for (int nbi = 0; nbi < 2; ++nbi) {
        const int n = (2 * w + nbi) * 16 + nl;
#pragma unroll
        for (int kk = 0; kk < 4; ++kk) {
            union { v16h v; v8h h[2]; } ub;
            const _Float16* p = wl + n * WS + kk * 32 + hi * 8;
            ub.h[0] = *(const v8h*)(p);      // K = kk*32 + hi*8 .. +7
            ub.h[1] = *(const v8h*)(p + 16); // K = kk*32 + 16 + hi*8 .. +7
            c[nbi] = __builtin_amdgcn_wmma_f32_16x16x32_f16(
                false, A[kk], false, ub.v, (short)0, c[nbi], false, false);
        }
    }
}

// Store this wave's two C/D fragments into ybuf as f16 row-major.
__device__ __forceinline__ void store_y(_Float16* __restrict__ yb,
                                        int lane, int w, const v8f vals[2]) {
    const int nl = lane & 15;
    const int hi = lane >> 4;
#pragma unroll
    for (int nbi = 0; nbi < 2; ++nbi) {
        const int n = (2 * w + nbi) * 16 + nl;
#pragma unroll
        for (int v = 0; v < 8; ++v) {
            yb[(v + 8 * hi) * YS + n] = (_Float16)vals[nbi][v];
        }
    }
}

// k = f(y) = silu(y@W1^T + b1) @ W2^T + b2 ; assumes ybuf holds y (post-barrier).
__device__ __forceinline__ void f_eval(Smem* sm, int lane, int w,
                                       float b1A, float b1B, float b2A, float b2B,
                                       v8f k[2]) {
    v8f z[2];
#pragma unroll
    for (int i = 0; i < 2; ++i) {
        const float bv = i ? b1B : b1A;
#pragma unroll
        for (int v = 0; v < 8; ++v) z[i][v] = bv;
    }
    gemm8(sm->ybuf, sm->w1, lane, w, z);
#pragma unroll
    for (int i = 0; i < 2; ++i)
#pragma unroll
        for (int v = 0; v < 8; ++v) z[i][v] = fast_silu(z[i][v]);
    __syncthreads();                    // all waves done reading y from ybuf
    store_y(sm->ybuf, lane, w, z);
    __syncthreads();
#pragma unroll
    for (int i = 0; i < 2; ++i) {
        const float bv = i ? b2B : b2A;
#pragma unroll
        for (int v = 0; v < 8; ++v) k[i][v] = bv;
    }
    gemm8(sm->ybuf, sm->w2, lane, w, k);
}

__global__ __launch_bounds__(128)
void rnnode_encoder_kernel(const float* __restrict__ x,     const float* __restrict__ span,
                           const float* __restrict__ W_emb, const float* __restrict__ b_emb,
                           const float* __restrict__ W_ih,  const float* __restrict__ b_ih,
                           const float* __restrict__ W_hh,  const float* __restrict__ b_hh,
                           const float* __restrict__ W1,    const float* __restrict__ b1,
                           const float* __restrict__ W2,    const float* __restrict__ b2,
                           const float* __restrict__ W_out, const float* __restrict__ b_out,
                           float* __restrict__ out) {
    __shared__ Smem sm;
    const int tid  = threadIdx.x;
    const int lane = tid & 31;
    const int w    = tid >> 5;        // wave id 0..3 (owns cols 32w..32w+31)
    const int b0   = blockIdx.x * 16; // batch tile

    // ---- init: weights -> LDS f16; fold embedding into (we, cc) ----
    for (int i = tid; i < H_SZ * H_SZ; i += 128) {
        const int r = i >> 7, c = i & 127;
        sm.whh[r * WS + c] = (_Float16)W_hh[i];
        sm.w1 [r * WS + c] = (_Float16)W1[i];
        sm.w2 [r * WS + c] = (_Float16)W2[i];
    }
    {
        float a = 0.0f, cacc = 0.0f;
        for (int d = 0; d < D_IN; ++d) {
            const float wv = W_ih[tid * D_IN + d];
            a    += wv * W_emb[d];
            cacc += wv * b_emb[d];
        }
        sm.we[tid] = a;
        sm.cc[tid] = cacc + b_ih[tid] + b_hh[tid];
    }
    const int nl = lane & 15;
    const int hi = lane >> 4;
    const int nA = 32 * w + nl;
    const int nB = nA + 16;
    const float b1A = b1[nA], b1B = b1[nB];
    const float b2A = b2[nA], b2B = b2[nB];
    __syncthreads();
    const float weA = sm.we[nA], weB = sm.we[nB];
    const float ccA = sm.cc[nA], ccB = sm.cc[nB];

    v8f h[2], k1[2], k2[2], k3[2], k4[2], y[2];
#pragma unroll
    for (int i = 0; i < 2; ++i)
#pragma unroll
        for (int v = 0; v < 8; ++v) h[i][v] = 0.0f;

    for (int t = 0; t < T_SZ - 1; ++t) {
        const float dt  = span[t + 1] - span[t];
        const float c13 = dt * (1.0f / 3.0f);
        const float c8  = dt * 0.125f;

        __syncthreads();                          // ybuf free from prev iteration
        store_y(sm.ybuf, lane, w, h);             // y = h (old) for recurrent GEMM
        if (tid < 16) sm.xbuf[tid] = x[(size_t)(b0 + tid) * T_SZ + t];
        __syncthreads();

        // ---- RNN jump: h = tanh(x*we + cc + h@W_hh^T) ----
        v8f pre[2];
#pragma unroll
        for (int i = 0; i < 2; ++i) {
            const float wev = i ? weB : weA;
            const float ccv = i ? ccB : ccA;
#pragma unroll
            for (int v = 0; v < 8; ++v)
                pre[i][v] = sm.xbuf[v + 8 * hi] * wev + ccv;
        }
        gemm8(sm.ybuf, sm.whh, lane, w, pre);
#pragma unroll
        for (int i = 0; i < 2; ++i)
#pragma unroll
            for (int v = 0; v < 8; ++v) h[i][v] = fast_tanh(pre[i][v]);

        // ---- RK4 (3/8 rule) ----
        // k1 = f(h)
        __syncthreads();
        store_y(sm.ybuf, lane, w, h);
        __syncthreads();
        f_eval(&sm, lane, w, b1A, b1B, b2A, b2B, k1);

        // k2 = f(h + dt/3 * k1)
#pragma unroll
        for (int i = 0; i < 2; ++i)
#pragma unroll
            for (int v = 0; v < 8; ++v) y[i][v] = h[i][v] + c13 * k1[i][v];
        __syncthreads();
        store_y(sm.ybuf, lane, w, y);
        __syncthreads();
        f_eval(&sm, lane, w, b1A, b1B, b2A, b2B, k2);

        // k3 = f(h + dt*k2 - dt/3*k1)
#pragma unroll
        for (int i = 0; i < 2; ++i)
#pragma unroll
            for (int v = 0; v < 8; ++v)
                y[i][v] = h[i][v] + dt * k2[i][v] - c13 * k1[i][v];
        __syncthreads();
        store_y(sm.ybuf, lane, w, y);
        __syncthreads();
        f_eval(&sm, lane, w, b1A, b1B, b2A, b2B, k3);

        // k4 = f(h + dt*(k1 - k2 + k3))
#pragma unroll
        for (int i = 0; i < 2; ++i)
#pragma unroll
            for (int v = 0; v < 8; ++v)
                y[i][v] = h[i][v] + dt * (k1[i][v] - k2[i][v] + k3[i][v]);
        __syncthreads();
        store_y(sm.ybuf, lane, w, y);
        __syncthreads();
        f_eval(&sm, lane, w, b1A, b1B, b2A, b2B, k4);

        // h += dt/8 * (k1 + 3*(k2 + k3) + k4)
#pragma unroll
        for (int i = 0; i < 2; ++i)
#pragma unroll
            for (int v = 0; v < 8; ++v)
                h[i][v] += c8 * (k1[i][v] + 3.0f * (k2[i][v] + k3[i][v]) + k4[i][v]);
    }

    // ---- epilogue: out = h @ W_out^T + b_out ----
    __syncthreads();
#pragma unroll
    for (int i = 0; i < 2; ++i) {
        const int n = (2 * w + i) * 16 + nl;
#pragma unroll
        for (int v = 0; v < 8; ++v)
            sm.hbuf[(v + 8 * hi) * H_SZ + n] = h[i][v];
    }
    __syncthreads();
    for (int o = tid * 8; o < tid * 8 + 8; ++o) {
        const int r = o >> 6;          // batch row in tile (0..15)
        const int c = o & 63;          // output column (0..63)
        float acc = b_out[c];
        for (int k = 0; k < H_SZ; ++k)
            acc += sm.hbuf[r * H_SZ + k] * W_out[c * H_SZ + k];
        out[(size_t)(b0 + r) * D_OUT + c] = acc;
    }
}

extern "C" void kernel_launch(void* const* d_in, const int* in_sizes, int n_in,
                              void* d_out, int out_size, void* d_ws, size_t ws_size,
                              hipStream_t stream) {
    (void)in_sizes; (void)n_in; (void)d_ws; (void)ws_size; (void)out_size;
    const float* x     = (const float*)d_in[0];
    const float* span  = (const float*)d_in[1];
    const float* W_emb = (const float*)d_in[2];
    const float* b_emb = (const float*)d_in[3];
    const float* W_ih  = (const float*)d_in[4];
    const float* b_ih  = (const float*)d_in[5];
    const float* W_hh  = (const float*)d_in[6];
    const float* b_hh  = (const float*)d_in[7];
    const float* W1    = (const float*)d_in[8];
    const float* b1    = (const float*)d_in[9];
    const float* W2    = (const float*)d_in[10];
    const float* b2    = (const float*)d_in[11];
    const float* W_out = (const float*)d_in[12];
    const float* b_out = (const float*)d_in[13];
    float* out = (float*)d_out;

    dim3 grid(B_SZ / 16);   // 128 workgroups, one 16-row batch tile each
    dim3 block(128);        // 4 wave32s: each owns a 32-column slice of H
    rnnode_encoder_kernel<<<grid, block, 0, stream>>>(
        x, span, W_emb, b_emb, W_ih, b_ih, W_hh, b_hh,
        W1, b1, W2, b2, W_out, b_out, out);
}
